// EfficientDet_66967130079727
// MI455X (gfx1250) — compile-verified, matrix operand order checked by reading, not run on hardware
//
#include <hip/hip_runtime.h>
#include <cstdint>
#include <cstddef>

typedef __bf16 bf16;
typedef __attribute__((ext_vector_type(16))) __bf16 v16bf;
typedef __attribute__((ext_vector_type(8)))  float  v8f;

#define CC 256
#define BN_EPS_F 1e-5f

// ---- CDNA5 async GMEM->LDS path (guarded; falls back to sync staging) ----
#if __has_builtin(__builtin_amdgcn_global_load_async_to_lds_b128) && \
    __has_builtin(__builtin_amdgcn_global_load_async_to_lds_b64)  && \
    __has_builtin(__builtin_amdgcn_s_wait_asynccnt)
#define USE_ASYNC_LDS 1
typedef int a_v4i __attribute__((vector_size(16)));
typedef int a_v2i __attribute__((vector_size(8)));
#define AS1 __attribute__((address_space(1)))
#define AS3 __attribute__((address_space(3)))
#define GP4(p) ((AS1 a_v4i*)(void*)(p))
#define LP4(p) ((AS3 a_v4i*)(void*)(p))
#define GP2(p) ((AS1 a_v2i*)(void*)(p))
#define LP2(p) ((AS3 a_v2i*)(void*)(p))
#else
#define USE_ASYNC_LDS 0
#endif

// -------------------------------------------------------------------------
// Fold conv-bias + BN into per-channel scale/shift.  ss[7][2][256]:
//   idx 0..4 : FPN levels (scale=1, shift=fpn_b)
//   idx 5    : cls head  (scale=inv, shift=bt+(b1-m)*inv)
//   idx 6    : reg head
// -------------------------------------------------------------------------
__global__ void ss_kernel(const float* __restrict__ fpn_b,
                          const float* __restrict__ cls_b1, const float* __restrict__ cg,
                          const float* __restrict__ cbt, const float* __restrict__ cm,
                          const float* __restrict__ cv,
                          const float* __restrict__ reg_b1, const float* __restrict__ rg,
                          const float* __restrict__ rbt, const float* __restrict__ rm,
                          const float* __restrict__ rv,
                          float* __restrict__ ss)
{
    int id = blockIdx.x * 256 + threadIdx.x;
    if (id >= 7 * 256) return;
    int s = id >> 8, c = id & 255;
    float scale, shift;
    if (s < 5)       { scale = 1.f; shift = fpn_b[s * 256 + c]; }
    else if (s == 5) { float inv = cg[c] * rsqrtf(cv[c] + BN_EPS_F);
                       scale = inv; shift = cbt[c] + (cls_b1[c] - cm[c]) * inv; }
    else             { float inv = rg[c] * rsqrtf(rv[c] + BN_EPS_F);
                       scale = inv; shift = rbt[c] + (reg_b1[c] - rm[c]) * inv; }
    ss[(s * 2 + 0) * 256 + c] = scale;
    ss[(s * 2 + 1) * 256 + c] = shift;
}

// -------------------------------------------------------------------------
// Repack 3x3 weights: OIHW fp32 -> [set][co][r*256+ci] bf16 (r = ky*3+kx).
// Sets: 0..4 = fpn_w levels, 5 = cls_w1, 6 = reg_w1.   589824 = 256*2304.
// -------------------------------------------------------------------------
__global__ void wconv_kernel(const float* __restrict__ fpn_w,
                             const float* __restrict__ cls_w1,
                             const float* __restrict__ reg_w1,
                             bf16* __restrict__ wb)
{
    size_t id = (size_t)blockIdx.x * blockDim.x + threadIdx.x;
    if (id >= (size_t)7 * 589824) return;
    int s  = (int)(id / 589824);
    int e  = (int)(id % 589824);
    int co = e / 2304;
    int k  = e % 2304;
    int r  = k >> 8;          // ky*3+kx
    int ci = k & 255;
    int ky = r / 3, kx = r % 3;
    const float* src = (s < 5) ? (fpn_w + (size_t)s * 589824)
                               : ((s == 5) ? cls_w1 : reg_w1);
    wb[id] = (bf16)src[(((size_t)co * 256 + ci) * 3 + ky) * 3 + kx];
}

// -------------------------------------------------------------------------
// Lateral 1x1 conv: NCHW fp32 input -> NHWC fp32 lateral.  Block = 1 pixel.
// -------------------------------------------------------------------------
__global__ __launch_bounds__(256) void lat_kernel(const float* __restrict__ f,
                                                  const float* __restrict__ w,
                                                  const float* __restrict__ bias,
                                                  float* __restrict__ latOut,
                                                  int Cin, int H, int W)
{
    __shared__ float xin[256];
    int p   = blockIdx.x;              // linear over (b,y,x)
    int hw  = H * W;
    int b   = p / hw, yx = p % hw;
    int tid = threadIdx.x;
    if (tid < Cin) xin[tid] = f[((size_t)b * Cin + tid) * hw + yx];
    __syncthreads();
    const float* wr = w + (size_t)tid * Cin;
    float s = bias[tid];
    for (int c = 0; c < Cin; ++c) s += xin[c] * wr[c];
    latOut[(size_t)p * CC + tid] = s;
}

// -------------------------------------------------------------------------
// Top-down: tgt[p][c] += src[b][y/2][x/2][c]   (nearest upsample ×2)
// -------------------------------------------------------------------------
__global__ void topdown_kernel(float* __restrict__ tgt, const float* __restrict__ src,
                               int Ht, int Wt, int Hs, int Ws, int total)
{
    int id = blockIdx.x * blockDim.x + threadIdx.x;
    if (id >= total) return;
    int c  = id & 255;
    int p  = id >> 8;
    int hw = Ht * Wt;
    int b  = p / hw, yx = p % hw;
    int y  = yx / Wt, x = yx % Wt;
    size_t sp = ((size_t)b * Hs + (y >> 1)) * Ws + (x >> 1);
    tgt[id] += src[sp * CC + c];
}

// fp32 -> bf16 elementwise
__global__ void f2b_kernel(const float* __restrict__ src, bf16* __restrict__ dst, size_t n)
{
    size_t id = (size_t)blockIdx.x * blockDim.x + threadIdx.x;
    if (id < n) dst[id] = (bf16)src[id];
}

// -------------------------------------------------------------------------
// Implicit-GEMM 3x3 conv, bf16 in / bf16 out, fp32 WMMA accumulation.
//   in  : [numPix][256] NHWC bf16 (numPix = B*H*W of this level)
//   wb  : [256][2304]   bf16, K-order (ky,kx,ci)
//   out : [numPix][256] NHWC bf16, y = relu?(acc*scale+shift)
// Block: 256 thr = 8 waves; tile 32 pixels x 256 co; 72 K-steps of 32.
// Wave w: pixel subtile (w&1), co subtiles 4*(w>>1)..+3 -> 4 wmma / step.
// Staging uses GLOBAL_LOAD_ASYNC_TO_LDS (ASYNCcnt) when available.
// -------------------------------------------------------------------------
__global__ __launch_bounds__(256) void conv3x3_wmma(const bf16* __restrict__ in,
                                                    const bf16* __restrict__ wb,
                                                    const float* __restrict__ scale,
                                                    const float* __restrict__ shift,
                                                    bf16* __restrict__ out,
                                                    int H, int W, int numPix, int relu)
{
    __shared__ __align__(16) bf16 As[CC][32];   // 16 KB  weights [co][k]
    __shared__ __align__(16) bf16 Bs[32][32];   //  2 KB  patch   [pix][k]

    const int tid    = threadIdx.x;
    const int wave   = tid >> 5;
    const int lane   = tid & 31;
    const int half   = lane >> 4;
    const int n16    = lane & 15;
    const int pixSub = wave & 1;
    const int coBase = (wave >> 1) * 64;

    const int pixBase = blockIdx.x * 32;

    // B-staging mapping: 8 threads per pixel, 4 contiguous ci each
    const int pi = tid >> 3;
    const int cq = (tid & 7) * 4;
    int p = pixBase + pi;
    bool pvalid = p < numPix;
    int pp  = pvalid ? p : 0;
    int hw  = H * W;
    int bimg = pp / hw;
    int rem  = pp % hw;
    int py = rem / W;
    int px = rem % W;

    v8f acc[4] = {};

    for (int kb = 0; kb < 72; ++kb) {
        int r   = kb >> 3;            // 0..8  -> (ky,kx)
        int ci0 = (kb & 7) * 32;
        int ky  = r / 3 - 1;
        int kx  = r % 3 - 1;

        const bf16* wsrc = wb + (size_t)tid * 2304 + kb * 32;   // 64B per co row
        int yy = py + ky, xx = px + kx;
        bool bvalid = pvalid && yy >= 0 && yy < H && xx >= 0 && xx < W;
        const bf16* bsrc = in + (((size_t)bimg * H + yy) * W + xx) * CC + ci0 + cq;

#if USE_ASYNC_LDS
        // ---- A: 64B per thread, direct GMEM->LDS (no VGPR bounce) ----
        __builtin_amdgcn_global_load_async_to_lds_b128(GP4(wsrc), LP4(&As[tid][0]),  0, 0);
        __builtin_amdgcn_global_load_async_to_lds_b128(GP4(wsrc), LP4(&As[tid][8]), 16, 0);
        __builtin_amdgcn_global_load_async_to_lds_b128(GP4(wsrc), LP4(&As[tid][16]),32, 0);
        __builtin_amdgcn_global_load_async_to_lds_b128(GP4(wsrc), LP4(&As[tid][24]),48, 0);
        // ---- B: 8B per thread; zero-fill padded lanes via plain ds store ----
        if (bvalid) {
            __builtin_amdgcn_global_load_async_to_lds_b64(GP2(bsrc), LP2(&Bs[pi][cq]), 0, 0);
        } else {
            uint2 z; z.x = 0u; z.y = 0u;
            *(uint2*)(&Bs[pi][cq]) = z;
        }
        __builtin_amdgcn_s_wait_asynccnt(0);
#else
        {   // fallback: distinct temporaries so the 4 loads can overlap
            const uint4* src4 = (const uint4*)wsrc;
            uint4 t0 = src4[0], t1 = src4[1], t2 = src4[2], t3 = src4[3];
            uint4* dst = (uint4*)(&As[tid][0]);
            dst[0] = t0; dst[1] = t1; dst[2] = t2; dst[3] = t3;
        }
        {
            uint2 v; v.x = 0u; v.y = 0u;
            if (bvalid) v = *(const uint2*)bsrc;
            *(uint2*)(&Bs[pi][cq]) = v;
        }
#endif
        if (kb + 1 < 72)   // next weight slice toward near cache
            __builtin_prefetch(wb + (size_t)tid * 2304 + (kb + 1) * 32, 0, 3);
        __syncthreads();

        // ---- B fragment (16-bit B 32x16 lane map: two 8-elem K runs) ----
        union { uint4 q[2]; v16bf v; } ub;
        {
            const bf16* bp = &Bs[pixSub * 16 + n16][0];
            ub.q[0] = *(const uint4*)(bp + half * 8);
            ub.q[1] = *(const uint4*)(bp + 16 + half * 8);
        }
        // ---- 4 A fragments + 4 WMMA ----
#pragma unroll
        for (int s = 0; s < 4; ++s) {
            union { uint4 q[2]; v16bf v; } ua;
            const bf16* ap = &As[coBase + s * 16 + n16][0];
            ua.q[0] = *(const uint4*)(ap + half * 8);
            ua.q[1] = *(const uint4*)(ap + 16 + half * 8);
            acc[s] = __builtin_amdgcn_wmma_f32_16x16x32_bf16(
                         false, ua.v, false, ub.v, (short)0, acc[s], false, false);
        }
        __syncthreads();
    }

    // ---- epilogue: scale/shift/relu, pack 8 bf16, one 16B store per tile ----
    int pout = pixBase + pixSub * 16 + n16;
    if (pout < numPix) {
#pragma unroll
        for (int s = 0; s < 4; ++s) {
            int co = coBase + s * 16 + half * 8;
            union { uint4 q; bf16 h[8]; } o;
#pragma unroll
            for (int r2 = 0; r2 < 8; ++r2) {
                float v = acc[s][r2] * scale[co + r2] + shift[co + r2];
                if (relu) v = v > 0.f ? v : 0.f;
                o.h[r2] = (bf16)v;
            }
            *(uint4*)(out + (size_t)pout * CC + co) = o.q;
        }
    }
}

// -------------------------------------------------------------------------
// Head 1x1 output conv, writes directly into [B, anchors, 14] layout.
//   hid : [numPix][256] bf16 (level slice)
//   nOut = A*dim (90 or 36), aDim = dim (10 or 4), dimOff = 4 (cls) / 0 (reg)
// -------------------------------------------------------------------------
__global__ __launch_bounds__(128) void head_out_kernel(const bf16* __restrict__ hid,
                                                       const float* __restrict__ w2,
                                                       const float* __restrict__ b2,
                                                       int nOut, int aDim, int dimOff,
                                                       int rowOff, int imgPix,
                                                       float* __restrict__ out)
{
    __shared__ __align__(4) bf16 h[CC];
    int p   = blockIdx.x;
    int tid = threadIdx.x;
    ((uint32_t*)h)[tid] = ((const uint32_t*)(hid + (size_t)p * CC))[tid];
    __syncthreads();
    if (tid < nOut) {
        int a = tid / aDim, j = tid % aDim;     // co = a*aDim + j
        const float* w = w2 + (size_t)tid * CC;
        float s = b2[tid];
        for (int c = 0; c < CC; ++c) s += (float)h[c] * w[c];
        int b  = p / imgPix;
        int yx = p % imgPix;
        size_t row = (size_t)b * 196416 + rowOff + (size_t)yx * 9 + a;
        out[row * 14 + dimOff + j] = s;
    }
}

// -------------------------------------------------------------------------
extern "C" void kernel_launch(void* const* d_in, const int* in_sizes, int n_in,
                              void* d_out, int out_size, void* d_ws, size_t ws_size,
                              hipStream_t stream)
{
    (void)in_sizes; (void)n_in; (void)out_size; (void)ws_size;

    const float* f[5];     for (int i = 0; i < 5; ++i) f[i]     = (const float*)d_in[i];
    const float* lat_w[5]; for (int i = 0; i < 5; ++i) lat_w[i] = (const float*)d_in[5 + i];
    const float* lat_b  = (const float*)d_in[10];
    const float* fpn_w  = (const float*)d_in[11];
    const float* fpn_b  = (const float*)d_in[12];
    const float* cls_w1 = (const float*)d_in[13];
    const float* cls_b1 = (const float*)d_in[14];
    const float* cls_g  = (const float*)d_in[15];
    const float* cls_bt = (const float*)d_in[16];
    const float* cls_m  = (const float*)d_in[17];
    const float* cls_v  = (const float*)d_in[18];
    const float* cls_w2 = (const float*)d_in[19];
    const float* cls_b2 = (const float*)d_in[20];
    const float* reg_w1 = (const float*)d_in[21];
    const float* reg_b1 = (const float*)d_in[22];
    const float* reg_g  = (const float*)d_in[23];
    const float* reg_bt = (const float*)d_in[24];
    const float* reg_m  = (const float*)d_in[25];
    const float* reg_v  = (const float*)d_in[26];
    const float* reg_w2 = (const float*)d_in[27];
    const float* reg_b2 = (const float*)d_in[28];
    float* out = (float*)d_out;

    static const int HWs[5]    = {128, 64, 32, 16, 8};
    static const int INCH[5]   = {32, 48, 96, 136, 232};
    static const int numPix[5] = {131072, 32768, 8192, 2048, 512};
    static const int pixOff[5] = {0, 131072, 163840, 172032, 174080};
    static const int rowOff[5] = {0, 147456, 184320, 193536, 195840};
    const size_t PIXTOT = 174592;

    // workspace carve-up (256B aligned)
    char*  base = (char*)d_ws;
    size_t off  = 0;
    auto take = [&](size_t bytes) -> char* {
        char* pp = base + off;
        off = (off + bytes + 255) & ~(size_t)255;
        return pp;
    };
    float* latF = (float*)take(PIXTOT * CC * sizeof(float));
    bf16*  latB = (bf16*) take(PIXTOT * CC * sizeof(bf16));
    bf16*  fpnB = (bf16*) take(PIXTOT * CC * sizeof(bf16));
    bf16*  hidB = (bf16*) take(PIXTOT * CC * sizeof(bf16));
    bf16*  wb   = (bf16*) take((size_t)7 * 589824 * sizeof(bf16));
    float* ss   = (float*)take(7 * 2 * 256 * sizeof(float));

    // 1) fold bias/BN
    ss_kernel<<<7, 256, 0, stream>>>(fpn_b, cls_b1, cls_g, cls_bt, cls_m, cls_v,
                                     reg_b1, reg_g, reg_bt, reg_m, reg_v, ss);
    // 2) weight repack fp32 -> bf16 K-major
    {
        size_t tot = (size_t)7 * 589824;
        wconv_kernel<<<(unsigned)((tot + 255) / 256), 256, 0, stream>>>(fpn_w, cls_w1, reg_w1, wb);
    }
    // 3) lateral 1x1 convs
    for (int l = 0; l < 5; ++l)
        lat_kernel<<<numPix[l], 256, 0, stream>>>(f[l], lat_w[l], lat_b + l * 256,
                                                  latF + (size_t)pixOff[l] * CC,
                                                  INCH[l], HWs[l], HWs[l]);
    // 4) top-down pathway (sequential: 4->3->2->1)
    for (int l = 3; l >= 0; --l) {
        int total = numPix[l] * CC;
        topdown_kernel<<<(total + 255) / 256, 256, 0, stream>>>(
            latF + (size_t)pixOff[l] * CC, latF + (size_t)pixOff[l + 1] * CC,
            HWs[l], HWs[l], HWs[l + 1], HWs[l + 1], total);
    }
    // 5) fp32 -> bf16
    {
        size_t n = PIXTOT * CC;
        f2b_kernel<<<(unsigned)((n + 255) / 256), 256, 0, stream>>>(latF, latB, n);
    }
    // 6) FPN 3x3 convs (bias only)
    for (int l = 0; l < 5; ++l)
        conv3x3_wmma<<<numPix[l] / 32, 256, 0, stream>>>(
            latB + (size_t)pixOff[l] * CC, wb + (size_t)l * 589824,
            ss + (l * 2) * 256, ss + (l * 2 + 1) * 256,
            fpnB + (size_t)pixOff[l] * CC, HWs[l], HWs[l], numPix[l], 0);
    // 7) cls head: 3x3 conv + BN + ReLU, then 1x1 -> out dims 4..13
    for (int l = 0; l < 5; ++l)
        conv3x3_wmma<<<numPix[l] / 32, 256, 0, stream>>>(
            fpnB + (size_t)pixOff[l] * CC, wb + (size_t)5 * 589824,
            ss + (5 * 2) * 256, ss + (5 * 2 + 1) * 256,
            hidB + (size_t)pixOff[l] * CC, HWs[l], HWs[l], numPix[l], 1);
    for (int l = 0; l < 5; ++l)
        head_out_kernel<<<numPix[l], 128, 0, stream>>>(
            hidB + (size_t)pixOff[l] * CC, cls_w2, cls_b2,
            90, 10, 4, rowOff[l], HWs[l] * HWs[l], out);
    // 8) reg head: 3x3 conv + BN + ReLU, then 1x1 -> out dims 0..3
    for (int l = 0; l < 5; ++l)
        conv3x3_wmma<<<numPix[l] / 32, 256, 0, stream>>>(
            fpnB + (size_t)pixOff[l] * CC, wb + (size_t)6 * 589824,
            ss + (6 * 2) * 256, ss + (6 * 2 + 1) * 256,
            hidB + (size_t)pixOff[l] * CC, HWs[l], HWs[l], numPix[l], 1);
    for (int l = 0; l < 5; ++l)
        head_out_kernel<<<numPix[l], 128, 0, stream>>>(
            hidB + (size_t)pixOff[l] * CC, reg_w2, reg_b2,
            36, 4, 0, rowOff[l], HWs[l] * HWs[l], out);
}
